// NKIMoEExpertMLP_33243046871379
// MI455X (gfx1250) — matrix operand order repaired and checked by reading, not compiled, not run on hardware
//
#include <hip/hip_runtime.h>

typedef __bf16 bf16;
typedef __attribute__((ext_vector_type(16))) __bf16 v16bf;
typedef __attribute__((ext_vector_type(8)))  __bf16 v8bf;
typedef __attribute__((ext_vector_type(4)))  __bf16 v4bf;
typedef __attribute__((ext_vector_type(8)))  float  v8f;

// T=2048, H=2048, I=768, E=16, TOPK=8
// LDS tile row stride: 40 bf16 = 80B (odd multiple of 16B -> minimal bank
// conflicts for ds_load_b128 fragment reads; rows stay 16B aligned).

// ---------------------------------------------------------------------------
// CDNA5 async global->LDS copy of 16 bytes (ASYNCcnt-tracked).
// ---------------------------------------------------------------------------
__device__ __forceinline__ void async_b128(unsigned lds_off, const void* base,
                                           int goff) {
  asm volatile("global_load_async_to_lds_b128 %0, %1, %2"
               :
               : "v"(lds_off), "v"(goff), "s"(base)
               : "memory");
}
__device__ __forceinline__ void wait_async0() {
  asm volatile("s_wait_asynccnt 0x0" ::: "memory");
}
__device__ __forceinline__ unsigned lds_addr(const void* p) {
  return (unsigned)(uintptr_t)p;  // low 32 bits of flat addr = LDS offset
}

// ---------------------------------------------------------------------------
// f32 -> bf16 convert (8 elements/thread)
// ---------------------------------------------------------------------------
__global__ __launch_bounds__(256) void cvt_x_kernel(const float* __restrict__ x,
                                                    bf16* __restrict__ y) {
  int idx = blockIdx.x * 256 + threadIdx.x;
  const float4* xv = (const float4*)x;
  float4 a = xv[idx * 2 + 0];
  float4 b = xv[idx * 2 + 1];
  bf16 o[8] = {(bf16)a.x, (bf16)a.y, (bf16)a.z, (bf16)a.w,
               (bf16)b.x, (bf16)b.y, (bf16)b.z, (bf16)b.w};
  *(v8bf*)(y + (size_t)idx * 8) = *(v8bf*)o;
}

// ---------------------------------------------------------------------------
// Per-expert transpose+convert: [e][R][C] f32 -> [e][C][R] bf16
// float4 loads, packed 8B bf16 stores.
// ---------------------------------------------------------------------------
__global__ __launch_bounds__(256) void transpose_cvt_kernel(
    const float* __restrict__ in, bf16* __restrict__ outp, int R, int C) {
  const int e = blockIdx.z;
  const float* ip = in + (size_t)e * R * C;
  bf16* op = outp + (size_t)e * R * C;
  const int r0 = blockIdx.y * 32, c0 = blockIdx.x * 32;
  __shared__ float tile[32][33];
  {
    int r = threadIdx.x >> 3, c4 = (threadIdx.x & 7) * 4;
    float4 v = *(const float4*)&ip[(size_t)(r0 + r) * C + c0 + c4];
    tile[r][c4 + 0] = v.x;
    tile[r][c4 + 1] = v.y;
    tile[r][c4 + 2] = v.z;
    tile[r][c4 + 3] = v.w;
  }
  __syncthreads();
  {
    int cl = threadIdx.x >> 3, rl = (threadIdx.x & 7) * 4;
    bf16 pk[4];
#pragma unroll
    for (int j = 0; j < 4; ++j) pk[j] = (bf16)tile[rl + j][cl];
    *(v4bf*)&op[(size_t)(c0 + cl) * R + r0 + rl] = *(v4bf*)pk;
  }
}

// ---------------------------------------------------------------------------
// Routing: combine[t][e] (duplicates summed) + per-expert token lists.
// counts[] must be zeroed beforehand.
// ---------------------------------------------------------------------------
__global__ __launch_bounds__(256) void routing_kernel(
    const int* __restrict__ eidx, const float* __restrict__ ew,
    float* __restrict__ combine, int* __restrict__ counts,
    int* __restrict__ tokList) {
  int t = blockIdx.x * 256 + threadIdx.x;
  int id[8];
  float w[8];
#pragma unroll
  for (int k = 0; k < 8; ++k) {
    id[k] = eidx[t * 8 + k];
    w[k] = ew[t * 8 + k];
  }
#pragma unroll
  for (int e = 0; e < 16; ++e) {
    float s = 0.0f;
#pragma unroll
    for (int k = 0; k < 8; ++k) s += (id[k] == e) ? w[k] : 0.0f;
    combine[t * 16 + e] = s;
    if (s > 0.0f) {
      int pos = atomicAdd(&counts[e], 1);
      tokList[e * 2048 + pos] = t;
    }
  }
}

// ---------------------------------------------------------------------------
// GEMM1 (token-gathered, sparse over assignments):
//   act[e][t][i] = combine[t][e] * silu(x@Wg) * (x@Wu)   (bf16)
// Double-buffered LDS, async global->LDS prefetch, 1 barrier per K-step.
// ---------------------------------------------------------------------------
__global__ __launch_bounds__(256) void gemm1_kernel(
    const bf16* __restrict__ X,        // [T][H] bf16
    const bf16* __restrict__ Wt,       // [E][2I][H] bf16 (transposed)
    const float* __restrict__ combine, // [T][E]
    const int* __restrict__ counts,    // [E]
    const int* __restrict__ tokList,   // [E][T]
    bf16* __restrict__ act)            // [E][T][I] bf16 (pre-zeroed)
{
  const int e = blockIdx.z;
  const int cnt = counts[e];
  const int tileBase = blockIdx.y * 128;
  if (tileBase >= cnt) return;  // uniform early-exit (sparse skip)
  const int nBase = blockIdx.x * 64;
  const int tid = threadIdx.x;
  const int lane = tid & 31;
  const int wave = tid >> 5;
  const int waveM = (wave & 3) * 32;
  const int waveN = (wave >> 2) * 32;
  const int l15 = lane & 15;
  const int g = lane >> 4;

  __shared__ bf16 As[2][128][40];
  __shared__ bf16 Bg[2][64][40];
  __shared__ bf16 Bu[2][64][40];
  __shared__ float Cs[8][16][17];

  v8f accG[2][2] = {};
  v8f accU[2][2] = {};

  const int lrow = tid >> 1;
  const int lhalf = (tid & 1) * 16;
  const int brow = (tid & 127) >> 1;
  const int bhalf = (tid & 1) * 16;
  const int rowSel = (tid < 128 ? nBase : 768 + nBase);

  int slotA = tileBase + lrow;
  if (slotA > cnt - 1) slotA = cnt - 1;
  const int tokA = tokList[e * 2048 + slotA];

  const int aBase = (tokA * 2048 + lhalf) * 2;                       // bytes
  const int bBase = ((e * 1536 + rowSel + brow) * 2048 + bhalf) * 2; // bytes

  unsigned laA[2], laB[2];
  laA[0] = lds_addr(&As[0][lrow][lhalf]);
  laA[1] = lds_addr(&As[1][lrow][lhalf]);
  laB[0] = (tid < 128) ? lds_addr(&Bg[0][brow][bhalf]) : lds_addr(&Bu[0][brow][bhalf]);
  laB[1] = (tid < 128) ? lds_addr(&Bg[1][brow][bhalf]) : lds_addr(&Bu[1][brow][bhalf]);

  async_b128(laA[0], X, aBase);
  async_b128(laA[0] + 16, X, aBase + 16);
  async_b128(laB[0], Wt, bBase);
  async_b128(laB[0] + 16, Wt, bBase + 16);

  for (int it = 0; it < 64; ++it) {
    const int cur = it & 1;
    wait_async0();
    __syncthreads();  // buf[cur] published; prior buffer reads all retired
    if (it + 1 < 64) {
      const int nk2 = (it + 1) * 64;  // k*2 bytes
      const int nxt = cur ^ 1;
      async_b128(laA[nxt], X, aBase + nk2);
      async_b128(laA[nxt] + 16, X, aBase + nk2 + 16);
      async_b128(laB[nxt], Wt, bBase + nk2);
      async_b128(laB[nxt] + 16, Wt, bBase + nk2 + 16);
    }
    v16bf af[2], bgf[2], buf2[2];
#pragma unroll
    for (int sm = 0; sm < 2; ++sm) {
      int r = waveM + sm * 16 + l15;
      ((v8bf*)&af[sm])[0] = *(const v8bf*)&As[cur][r][g * 16];
      ((v8bf*)&af[sm])[1] = *(const v8bf*)&As[cur][r][g * 16 + 8];
    }
#pragma unroll
    for (int sn = 0; sn < 2; ++sn) {
      int r = waveN + sn * 16 + l15;
      ((v8bf*)&bgf[sn])[0] = *(const v8bf*)&Bg[cur][r][g * 16];
      ((v8bf*)&bgf[sn])[1] = *(const v8bf*)&Bg[cur][r][g * 16 + 8];
      ((v8bf*)&buf2[sn])[0] = *(const v8bf*)&Bu[cur][r][g * 16];
      ((v8bf*)&buf2[sn])[1] = *(const v8bf*)&Bu[cur][r][g * 16 + 8];
    }
#pragma unroll
    for (int sm = 0; sm < 2; ++sm)
#pragma unroll
      for (int sn = 0; sn < 2; ++sn) {
        accG[sm][sn] = __builtin_amdgcn_wmma_f32_16x16x32_bf16(
            false, af[sm], false, bgf[sn], (short)0, accG[sm][sn], false, false);
        accU[sm][sn] = __builtin_amdgcn_wmma_f32_16x16x32_bf16(
            false, af[sm], false, buf2[sn], (short)0, accU[sm][sn], false, false);
      }
  }

  float(*cs)[17] = Cs[wave];
  const int orow = lane >> 1;
  const int ohalf = (lane & 1) * 8;
#pragma unroll
  for (int sm = 0; sm < 2; ++sm)
#pragma unroll
    for (int sn = 0; sn < 2; ++sn) {
#pragma unroll
      for (int r = 0; r < 8; ++r) {
        float gt = accG[sm][sn][r];
        float up = accU[sm][sn][r];
        float sl = gt * __builtin_amdgcn_rcpf(1.0f + __expf(-gt));
        cs[r + 8 * g][l15] = sl * up;
      }
      int slot = tileBase + waveM + sm * 16 + orow;
      if (slot < cnt) {  // suppress clamped (padding) rows
        int t = tokList[e * 2048 + slot];
        int ic = nBase + waveN + sn * 16 + ohalf;
        float w = combine[t * 16 + e];
        bf16 pk[8];
#pragma unroll
        for (int j = 0; j < 8; ++j) pk[j] = (bf16)(cs[orow][ohalf + j] * w);
        *(v8bf*)&act[((size_t)e * 2048 + t) * 768 + ic] = *(v8bf*)pk;
      }
    }
}

// ---------------------------------------------------------------------------
// GEMM2: out[t][h] = sum_e act[e][t][:] @ down_e  (down stored [e][h][i])
// Dense K = 16*768, double-buffered LDS with async prefetch.
// ---------------------------------------------------------------------------
__global__ __launch_bounds__(256) void gemm2_kernel(
    const bf16* __restrict__ act,  // [E][T][I]
    const bf16* __restrict__ Dt,   // [E][H][I]
    float* __restrict__ out)       // [T][H]
{
  const int mBase = blockIdx.y * 128;
  const int nBase = blockIdx.x * 128;
  const int tid = threadIdx.x;
  const int lane = tid & 31;
  const int wave = tid >> 5;
  const int waveM = (wave & 1) * 64;
  const int waveN = (wave >> 1) * 32;
  const int l15 = lane & 15;
  const int g = lane >> 4;

  __shared__ bf16 As[2][128][40];
  __shared__ bf16 Bs[2][128][40];
  __shared__ float Cs[8][16][17];

  v8f acc[4][2] = {};
  const int lrow = tid >> 1;
  const int lhalf = (tid & 1) * 16;

  const int aRow = ((mBase + lrow) * 768 + lhalf) * 2;
  const int bRow = ((nBase + lrow) * 768 + lhalf) * 2;
  const int eStride = 2048 * 768 * 2;

  unsigned laA[2], laB[2];
  laA[0] = lds_addr(&As[0][lrow][lhalf]);
  laA[1] = lds_addr(&As[1][lrow][lhalf]);
  laB[0] = lds_addr(&Bs[0][lrow][lhalf]);
  laB[1] = lds_addr(&Bs[1][lrow][lhalf]);

  async_b128(laA[0], act, aRow);
  async_b128(laA[0] + 16, act, aRow + 16);
  async_b128(laB[0], Dt, bRow);
  async_b128(laB[0] + 16, Dt, bRow + 16);

  for (int it = 0; it < 384; ++it) {  // e = it/24, k = (it%24)*32
    const int cur = it & 1;
    wait_async0();
    __syncthreads();
    if (it + 1 < 384) {
      const int it1 = it + 1;
      const int e1 = it1 / 24;
      const int k2 = (it1 - e1 * 24) * 64;
      const int nxt = cur ^ 1;
      const int ga = e1 * eStride + aRow + k2;
      const int gb = e1 * eStride + bRow + k2;
      async_b128(laA[nxt], act, ga);
      async_b128(laA[nxt] + 16, act, ga + 16);
      async_b128(laB[nxt], Dt, gb);
      async_b128(laB[nxt] + 16, Dt, gb + 16);
    }
    v16bf af[4], bf2[2];
#pragma unroll
    for (int sm = 0; sm < 4; ++sm) {
      int r = waveM + sm * 16 + l15;
      ((v8bf*)&af[sm])[0] = *(const v8bf*)&As[cur][r][g * 16];
      ((v8bf*)&af[sm])[1] = *(const v8bf*)&As[cur][r][g * 16 + 8];
    }
#pragma unroll
    for (int sn = 0; sn < 2; ++sn) {
      int r = waveN + sn * 16 + l15;
      ((v8bf*)&bf2[sn])[0] = *(const v8bf*)&Bs[cur][r][g * 16];
      ((v8bf*)&bf2[sn])[1] = *(const v8bf*)&Bs[cur][r][g * 16 + 8];
    }
#pragma unroll
    for (int sm = 0; sm < 4; ++sm)
#pragma unroll
      for (int sn = 0; sn < 2; ++sn)
        acc[sm][sn] = __builtin_amdgcn_wmma_f32_16x16x32_bf16(
            false, af[sm], false, bf2[sn], (short)0, acc[sm][sn], false, false);
  }

  float(*cs)[17] = Cs[wave];
  const int orow = lane >> 1;
  const int ohalf = (lane & 1) * 8;
#pragma unroll
  for (int sm = 0; sm < 4; ++sm)
#pragma unroll
    for (int sn = 0; sn < 2; ++sn) {
#pragma unroll
      for (int r = 0; r < 8; ++r) cs[r + 8 * g][l15] = acc[sm][sn][r];
      int t = mBase + waveM + sm * 16 + orow;
      int h = nBase + waveN + sn * 16 + ohalf;
      float4 o0, o1;
      o0.x = cs[orow][ohalf + 0];
      o0.y = cs[orow][ohalf + 1];
      o0.z = cs[orow][ohalf + 2];
      o0.w = cs[orow][ohalf + 3];
      o1.x = cs[orow][ohalf + 4];
      o1.y = cs[orow][ohalf + 5];
      o1.z = cs[orow][ohalf + 6];
      o1.w = cs[orow][ohalf + 7];
      *(float4*)&out[(size_t)t * 2048 + h] = o0;
      *(float4*)&out[(size_t)t * 2048 + h + 4] = o1;
    }
}

// ---------------------------------------------------------------------------
// Launch
// ---------------------------------------------------------------------------
extern "C" void kernel_launch(void* const* d_in, const int* in_sizes, int n_in,
                              void* d_out, int out_size, void* d_ws,
                              size_t ws_size, hipStream_t stream) {
  (void)in_sizes; (void)n_in; (void)out_size; (void)ws_size;
  const float* hs = (const float*)d_in[0];    // [2048][2048]
  const float* gup = (const float*)d_in[1];   // [16][2048][1536]
  const float* down = (const float*)d_in[2];  // [16][768][2048]
  const int* eidx = (const int*)d_in[3];      // [2048][8]
  const float* ew = (const float*)d_in[4];    // [2048][8]
  float* out = (float*)d_out;                 // [2048][2048]

  char* ws = (char*)d_ws;
  size_t off = 0;
  bf16* Xbf = (bf16*)(ws + off);   off += (size_t)2048 * 2048 * 2;
  bf16* gupT = (bf16*)(ws + off);  off += (size_t)16 * 1536 * 2048 * 2;
  bf16* dwnT = (bf16*)(ws + off);  off += (size_t)16 * 2048 * 768 * 2;
  bf16* actw = (bf16*)(ws + off);
  const size_t actBytes = (size_t)16 * 2048 * 768 * 2;
  off += actBytes;
  float* comb = (float*)(ws + off); off += (size_t)2048 * 16 * 4;
  int* counts = (int*)(ws + off);   off += 16 * sizeof(int);
  int* tokList = (int*)(ws + off);  off += (size_t)16 * 2048 * sizeof(int);

  cvt_x_kernel<<<2048, 256, 0, stream>>>(hs, Xbf);
  transpose_cvt_kernel<<<dim3(48, 64, 16), 256, 0, stream>>>(gup, gupT, 2048, 1536);
  transpose_cvt_kernel<<<dim3(64, 24, 16), 256, 0, stream>>>(down, dwnT, 768, 2048);
  hipMemsetAsync(counts, 0, 16 * sizeof(int), stream);
  hipMemsetAsync(actw, 0, actBytes, stream);  // unassigned (t,e) rows stay 0
  routing_kernel<<<8, 256, 0, stream>>>(eidx, ew, comb, counts, tokList);
  gemm1_kernel<<<dim3(12, 16, 16), 256, 0, stream>>>(Xbf, gupT, comb, counts,
                                                     tokList, actw);
  gemm2_kernel<<<dim3(16, 16), 256, 0, stream>>>(actw, dwnT, out);
}
